// MultiHeadAttention_13185549598915
// MI455X (gfx1250) — compile-verified, matrix operand order checked by reading, not run on hardware
//
#include <hip/hip_runtime.h>
#include <hip/hip_bf16.h>
#include <math.h>

#define BATCH 4
#define SEQ 2048
#define DMODEL 1024
#define HEADS 16
#define HDIM 64

typedef __attribute__((ext_vector_type(8)))  __bf16 v8bf;
typedef __attribute__((ext_vector_type(16))) __bf16 v16bf;
typedef __attribute__((ext_vector_type(8)))  float  v8f;

#define WMMA_BF16(A, B, C) \
    __builtin_amdgcn_wmma_f32_16x16x32_bf16(false, (A), false, (B), (short)0, (C), false, false)

// ---------------------------------------------------------------------------
// Fragment load: 16-bit A/B fragment for v_wmma_*_16x16x32_bf16.
// Lane l (l<16)  holds row/col (base+l),    K in {0..7, 16..23}
// Lane l (l>=16) holds row/col (base+l-16), K in {8..15, 24..31}
// i.e. two contiguous 8-element (16B) chunks at k0 + half*8 and k0 + 16 + half*8.
// ---------------------------------------------------------------------------
__device__ __forceinline__ v16bf load_frag(const __bf16* base, int ld, int row,
                                           int k0, int lane) {
    const int half = lane >> 4;
    const int ln   = lane & 15;
    const __bf16* p = base + (size_t)(row + ln) * ld + k0 + half * 8;
    v8bf lo = *(const v8bf*)(p);
    v8bf hi = *(const v8bf*)(p + 16);
    v16bf f;
#pragma unroll
    for (int i = 0; i < 8; ++i) { f[i] = lo[i]; f[i + 8] = hi[i]; }
    return f;
}

// ---------------------------------------------------------------------------
// fp32 -> bf16 conversion, 8 elements per thread (b128 in, b128 out)
// ---------------------------------------------------------------------------
__global__ void cvt_bf16_kernel(const float* __restrict__ in,
                                __bf16* __restrict__ out, int n) {
    const int i = (blockIdx.x * blockDim.x + threadIdx.x) * 8;
    if (i < n) {
        const float4 a = *(const float4*)(in + i);
        const float4 b = *(const float4*)(in + i + 4);
        v8bf o;
        o[0] = (__bf16)a.x; o[1] = (__bf16)a.y; o[2] = (__bf16)a.z; o[3] = (__bf16)a.w;
        o[4] = (__bf16)b.x; o[5] = (__bf16)b.y; o[6] = (__bf16)b.z; o[7] = (__bf16)b.w;
        *(v8bf*)(out + i) = o;
    }
}

// ---------------------------------------------------------------------------
// QKV projection + fused RoPE.
//   X  : [B*S, D]  bf16 (row-major, K contiguous)
//   W  : [3D, D]   bf16 (row-major, K contiguous)  -> acts as B^T columns
//   Qb : [B,H,S,hd] bf16  (RoPE'd, pre-scaled by 1/sqrt(hd))
//   Kb : [B,H,S,hd] bf16  (RoPE'd)
//   Vt : [B,H,hd,S] bf16  (transposed for PV B-fragments)
// Each wave: 64x64 output tile, 16 WMMAs per K-step of 32.
// ---------------------------------------------------------------------------
__global__ __launch_bounds__(256) void qkv_rope_kernel(
    const __bf16* __restrict__ X, const __bf16* __restrict__ W,
    __bf16* __restrict__ Qb, __bf16* __restrict__ Kb, __bf16* __restrict__ Vt) {
    const int lane = threadIdx.x & 31;
    const int wid  = threadIdx.x >> 5;
    const int tile = blockIdx.x * 8 + wid;
    const int NT = (3 * DMODEL) / 64;          // 48
    const int tm = tile / NT, tn = tile % NT;
    const int m0 = tm * 64, n0 = tn * 64;
    const int half = lane >> 4, ln = lane & 15;

    v8f acc[4][4] = {};
    for (int k0 = 0; k0 < DMODEL; k0 += 32) {
        // prefetch two K-steps ahead (global_prefetch_b8), clamped in-buffer
        const int kpf = (k0 + 64 < DMODEL) ? k0 + 64 : DMODEL - 32;
        __builtin_prefetch(X + (size_t)(m0 + ln) * DMODEL + kpf + half * 8, 0, 1);
        __builtin_prefetch(W + (size_t)(n0 + ln) * DMODEL + kpf + half * 8, 0, 1);

        v16bf af[4], bfr[4];
#pragma unroll
        for (int i = 0; i < 4; ++i) af[i]  = load_frag(X, DMODEL, m0 + 16 * i, k0, lane);
#pragma unroll
        for (int j = 0; j < 4; ++j) bfr[j] = load_frag(W, DMODEL, n0 + 16 * j, k0, lane);
#pragma unroll
        for (int i = 0; i < 4; ++i)
#pragma unroll
            for (int j = 0; j < 4; ++j)
                acc[i][j] = WMMA_BF16(af[i], bfr[j], acc[i][j]);
    }

    // Epilogue. C-fragment: element e of tile (i,j) lives at
    //   row = m0+16*i + e + 8*(lane>>4),  col = n0+16*j + (lane&15)
    const int e0 = n0;                          // global feature of this 64-wide tile
    if (e0 < 2 * DMODEL) {
        // -------- Q or K head: apply rotate-half RoPE --------
        const bool isQ = (e0 < DMODEL);
        const int  h   = (isQ ? e0 : e0 - DMODEL) >> 6;
        __bf16* dst    = isQ ? Qb : Kb;
        const float scale = isQ ? 0.125f : 1.0f;   // 1/sqrt(64) folded into Q
#pragma unroll
        for (int j = 0; j < 2; ++j) {
            const int c1 = 16 * j + ln;            // 0..31, freq index
            // inv_freq[c1] = 10000^(-c1/32) = exp2(-c1 * log2(10000)/32)
            const float invf = exp2f(-(float)c1 * 0.4152410118609203f);
#pragma unroll
            for (int i = 0; i < 4; ++i) {
#pragma unroll
                for (int e = 0; e < 8; ++e) {
                    const int sidx = m0 + 16 * i + e + 8 * half; // b*S+s
                    const int b = sidx >> 11, s = sidx & (SEQ - 1);
                    float sn, cs;
                    __sincosf((float)s * invf, &sn, &cs);
                    const float x1 = acc[i][j][e];
                    const float x2 = acc[i][j + 2][e];
                    const float o1 = (x1 * cs - x2 * sn) * scale;
                    const float o2 = (x2 * cs + x1 * sn) * scale;
                    const size_t base = ((size_t)(b * HEADS + h) * SEQ + s) * HDIM;
                    dst[base + c1]      = (__bf16)o1;
                    dst[base + c1 + 32] = (__bf16)o2;
                }
            }
        }
    } else {
        // -------- V head: store transposed [b,h,d,s] --------
        const int h = (e0 - 2 * DMODEL) >> 6;
#pragma unroll
        for (int j = 0; j < 4; ++j) {
            const int d = 16 * j + ln;
#pragma unroll
            for (int i = 0; i < 4; ++i) {
#pragma unroll
                for (int e = 0; e < 8; ++e) {
                    const int sidx = m0 + 16 * i + e + 8 * half;
                    const int b = sidx >> 11, s = sidx & (SEQ - 1);
                    Vt[((size_t)(b * HEADS + h) * HDIM + d) * SEQ + s] =
                        (__bf16)acc[i][j][e];
                }
            }
        }
    }
}

// ---------------------------------------------------------------------------
// Causal flash attention. One wave = 32 queries x one (b,h) head.
// Per 32-key block: 8 WMMA for QK^T (K fragments reused across both query
// sub-tiles), online softmax in registers (row stats replicated across each
// 16-lane half), P -> LDS -> A-fragments, 8 WMMA for P*V (V fragments reused).
// Output bf16 [B,S,D].
// ---------------------------------------------------------------------------
__global__ __launch_bounds__(256) void flash_attn_kernel(
    const __bf16* __restrict__ Qb, const __bf16* __restrict__ Kb,
    const __bf16* __restrict__ Vt, __bf16* __restrict__ O) {
    __shared__ __align__(16) __bf16 plds[8][32 * 32];   // 2KB per wave

    const int lane = threadIdx.x & 31;
    const int wid  = threadIdx.x >> 5;
    const int tile = blockIdx.x * 8 + wid;
    const int bh   = tile >> 6;                 // (b*H + h), S/32 = 64 tiles/head
    const int q0   = (tile & 63) * 32;
    const int half = lane >> 4, ln = lane & 15;

    const __bf16* Qp = Qb + (size_t)bh * SEQ * HDIM;
    const __bf16* Kp = Kb + (size_t)bh * SEQ * HDIM;
    const __bf16* Vp = Vt + (size_t)bh * HDIM * SEQ;

    v16bf qf[2][2];
#pragma unroll
    for (int r = 0; r < 2; ++r)
#pragma unroll
        for (int kk = 0; kk < 2; ++kk)
            qf[r][kk] = load_frag(Qp, HDIM, q0 + 16 * r, kk * 32, lane);

    v8f acc[2][4] = {};
    float mrun[2][8], lrun[2][8];
#pragma unroll
    for (int r = 0; r < 2; ++r)
#pragma unroll
        for (int e = 0; e < 8; ++e) { mrun[r][e] = -1e30f; lrun[r][e] = 0.0f; }

    __bf16* pw = plds[wid];
    constexpr float LOG2E = 1.4426950408889634f;

    for (int kb = 0; kb < q0 + 32; kb += 32) {
        // ---- scores: 32 queries x 32 keys ----
        v8f s[2][2] = {};
#pragma unroll
        for (int kk = 0; kk < 2; ++kk) {
            v16bf kf0 = load_frag(Kp, HDIM, kb,      kk * 32, lane);
            v16bf kf1 = load_frag(Kp, HDIM, kb + 16, kk * 32, lane);
#pragma unroll
            for (int r = 0; r < 2; ++r) {
                s[r][0] = WMMA_BF16(qf[r][kk], kf0, s[r][0]);
                s[r][1] = WMMA_BF16(qf[r][kk], kf1, s[r][1]);
            }
        }
        // ---- causal mask (only in diagonal region) ----
        if (kb + 31 > q0) {
#pragma unroll
            for (int r = 0; r < 2; ++r)
#pragma unroll
                for (int e = 0; e < 8; ++e) {
                    const int qrow = q0 + 16 * r + e + 8 * half;
                    if (kb + ln      > qrow) s[r][0][e] = -1e30f;
                    if (kb + 16 + ln > qrow) s[r][1][e] = -1e30f;
                }
        }
        // ---- online softmax + P -> LDS ----
#pragma unroll
        for (int r = 0; r < 2; ++r) {
            float corr[8];
#pragma unroll
            for (int e = 0; e < 8; ++e) {
                float mx = fmaxf(s[r][0][e], s[r][1][e]);
#pragma unroll
                for (int m = 1; m < 16; m <<= 1) mx = fmaxf(mx, __shfl_xor(mx, m));
                const float mnew = fmaxf(mrun[r][e], mx);
                const float p0 = exp2f((s[r][0][e] - mnew) * LOG2E);
                const float p1 = exp2f((s[r][1][e] - mnew) * LOG2E);
                float ps = p0 + p1;
#pragma unroll
                for (int m = 1; m < 16; m <<= 1) ps += __shfl_xor(ps, m);
                corr[e]    = exp2f((mrun[r][e] - mnew) * LOG2E);
                lrun[r][e] = lrun[r][e] * corr[e] + ps;
                mrun[r][e] = mnew;
                const int row = 16 * r + e + 8 * half;
                pw[row * 32 + ln]      = (__bf16)p0;
                pw[row * 32 + 16 + ln] = (__bf16)p1;
            }
#pragma unroll
            for (int j = 0; j < 4; ++j)
#pragma unroll
                for (int e = 0; e < 8; ++e) acc[r][j][e] *= corr[e];
        }
        asm volatile("" ::: "memory");          // same-wave LDS ops are in-order
        v16bf pf[2];
        pf[0] = load_frag(pw, 32, 0,  0, lane);
        pf[1] = load_frag(pw, 32, 16, 0, lane);
        asm volatile("" ::: "memory");

        // ---- acc += P * V  (V fragment shared by both query sub-tiles) ----
#pragma unroll
        for (int j = 0; j < 4; ++j) {
            v16bf vf = load_frag(Vp, SEQ, 16 * j, kb, lane);
#pragma unroll
            for (int r = 0; r < 2; ++r)
                acc[r][j] = WMMA_BF16(pf[r], vf, acc[r][j]);
        }
    }

    // ---- normalize and store [B,S,D] ----
    const int b = bh >> 4, h = bh & 15;
#pragma unroll
    for (int r = 0; r < 2; ++r)
#pragma unroll
        for (int j = 0; j < 4; ++j) {
            const int d = 16 * j + ln;
#pragma unroll
            for (int e = 0; e < 8; ++e) {
                const int q = q0 + 16 * r + e + 8 * half;
                const float o = acc[r][j][e] / lrun[r][e];
                O[((size_t)(b * SEQ + q)) * DMODEL + h * HDIM + d] = (__bf16)o;
            }
        }
}

// ---------------------------------------------------------------------------
// Output projection: out[M=8192, N=1024] = A[M,K=1024] * W^T, fp32 result.
// ---------------------------------------------------------------------------
__global__ __launch_bounds__(256) void outproj_kernel(
    const __bf16* __restrict__ A, const __bf16* __restrict__ W,
    float* __restrict__ out) {
    const int lane = threadIdx.x & 31;
    const int wid  = threadIdx.x >> 5;
    const int tile = blockIdx.x * 8 + wid;
    const int NT = DMODEL / 64;                 // 16
    const int tm = tile / NT, tn = tile % NT;
    const int m0 = tm * 64, n0 = tn * 64;
    const int half = lane >> 4, ln = lane & 15;

    v8f acc[4][4] = {};
    for (int k0 = 0; k0 < DMODEL; k0 += 32) {
        const int kpf = (k0 + 64 < DMODEL) ? k0 + 64 : DMODEL - 32;
        __builtin_prefetch(A + (size_t)(m0 + ln) * DMODEL + kpf + half * 8, 0, 1);
        __builtin_prefetch(W + (size_t)(n0 + ln) * DMODEL + kpf + half * 8, 0, 1);

        v16bf af[4], bfr[4];
#pragma unroll
        for (int i = 0; i < 4; ++i) af[i]  = load_frag(A, DMODEL, m0 + 16 * i, k0, lane);
#pragma unroll
        for (int j = 0; j < 4; ++j) bfr[j] = load_frag(W, DMODEL, n0 + 16 * j, k0, lane);
#pragma unroll
        for (int i = 0; i < 4; ++i)
#pragma unroll
            for (int j = 0; j < 4; ++j)
                acc[i][j] = WMMA_BF16(af[i], bfr[j], acc[i][j]);
    }

#pragma unroll
    for (int i = 0; i < 4; ++i)
#pragma unroll
        for (int j = 0; j < 4; ++j)
#pragma unroll
            for (int e = 0; e < 8; ++e) {
                const int row = m0 + 16 * i + e + 8 * half;
                const int col = n0 + 16 * j + ln;
                out[(size_t)row * DMODEL + col] = acc[i][j][e];
            }
}

// ---------------------------------------------------------------------------
// Host launcher
// ---------------------------------------------------------------------------
extern "C" void kernel_launch(void* const* d_in, const int* in_sizes, int n_in,
                              void* d_out, int out_size, void* d_ws, size_t ws_size,
                              hipStream_t stream) {
    (void)in_sizes; (void)n_in; (void)out_size; (void)ws_size;
    const float* x     = (const float*)d_in[0];   // [4,2048,1024]
    const float* w_qkv = (const float*)d_in[1];   // [3072,1024]
    const float* w_out = (const float*)d_in[2];   // [1024,1024]
    // d_in[3] = causal mask, implicit in the kernel.
    float* out = (float*)d_out;

    const int nX  = BATCH * SEQ * DMODEL;         // 8388608
    const int nWq = 3 * DMODEL * DMODEL;          // 3145728
    const int nWo = DMODEL * DMODEL;              // 1048576

    char* ws = (char*)d_ws;
    __bf16* xb    = (__bf16*)ws; ws += (size_t)nX  * 2;
    __bf16* wqkvb = (__bf16*)ws; ws += (size_t)nWq * 2;
    __bf16* woutb = (__bf16*)ws; ws += (size_t)nWo * 2;
    __bf16* Qb    = (__bf16*)ws; ws += (size_t)nX  * 2;
    __bf16* Kb    = (__bf16*)ws; ws += (size_t)nX  * 2;
    __bf16* Vt    = (__bf16*)ws; ws += (size_t)nX  * 2;
    __bf16* Ab    = (__bf16*)ws; ws += (size_t)nX  * 2;

    cvt_bf16_kernel<<<(nX  / 8 + 255) / 256, 256, 0, stream>>>(x, xb, nX);
    cvt_bf16_kernel<<<(nWq / 8 + 255) / 256, 256, 0, stream>>>(w_qkv, wqkvb, nWq);
    cvt_bf16_kernel<<<(nWo / 8 + 255) / 256, 256, 0, stream>>>(w_out, woutb, nWo);

    // QKV + RoPE: (8192/64) * (3072/64) = 6144 wave-tiles, 8 waves/block
    qkv_rope_kernel<<<6144 / 8, 256, 0, stream>>>(xb, wqkvb, Qb, Kb, Vt);

    // Flash attention: B*H*(S/32) = 4096 waves, 8 waves/block
    flash_attn_kernel<<<4096 / 8, 256, 0, stream>>>(Qb, Kb, Vt, Ab);

    // Output projection: (8192/64) * (1024/64) = 2048 wave-tiles
    outproj_kernel<<<2048 / 8, 256, 0, stream>>>(Ab, woutb, out);
}
